// A3TGCN_2_points_53566832115798
// MI455X (gfx1250) — matrix operand order repaired and checked by reading, not compiled, not run on hardware
//
#include <hip/hip_runtime.h>
#include <math.h>

// ---------------------------------------------------------------------------
// Problem constants (from the reference)
// ---------------------------------------------------------------------------
#define BB   512
#define CC   64
#define NN   32
#define VV   100
#define EMB  32
#define HID  64
#define EE   256
#define TT   37

typedef __attribute__((ext_vector_type(16))) _Float16 v16h;
typedef __attribute__((ext_vector_type(8)))  float    v8f;

// ---------------------------------------------------------------------------
// Workspace layout (all precomputed by setup kernel):
//   _Float16 A[32*32]       : normalized adjacency (dense, incl. self loops), row-major
//   _Float16 Mt[128*32]     : TRANSPOSED [W_z@Lz_top | W_h@Lh_top]; Mt[col][k], K contiguous
//   float    cz[64], ch[64] : folded biases  b@L_top + lb
//   float    prefix[38]     : prefix sums of softmax(att); Pa = prefix[LOS]
// ---------------------------------------------------------------------------

__global__ __launch_bounds__(128)
void tgcn_setup_kernel(const int* __restrict__ tei,
                       const float* __restrict__ W_z, const float* __restrict__ b_z,
                       const float* __restrict__ W_h, const float* __restrict__ b_h,
                       const float* __restrict__ L_z, const float* __restrict__ lb_z,
                       const float* __restrict__ L_h, const float* __restrict__ lb_h,
                       const float* __restrict__ att,
                       _Float16* __restrict__ wsA, _Float16* __restrict__ wsMt,
                       float* __restrict__ wscz, float* __restrict__ wsch,
                       float* __restrict__ wsprefix) {
  __shared__ float deg[NN];
  __shared__ float dinv[NN];
  __shared__ float Af[NN * NN];
  const int tid = threadIdx.x;

  for (int i = tid; i < NN; i += 128) deg[i] = 0.f;
  for (int i = tid; i < NN * NN; i += 128) Af[i] = 0.f;
  __syncthreads();

  // degree over dst (E template edges + N self loops)
  for (int e = tid; e < EE + NN; e += 128) {
    int d = (e < EE) ? tei[EE + e] : (e - EE);
    atomicAdd(&deg[d], 1.f);
  }
  __syncthreads();
  for (int i = tid; i < NN; i += 128)
    dinv[i] = (deg[i] > 0.f) ? (1.0f / sqrtf(deg[i])) : 0.f;
  __syncthreads();
  // A[dst][src] += dinv[src]*dinv[dst]
  for (int e = tid; e < EE + NN; e += 128) {
    int s = (e < EE) ? tei[e]      : (e - EE);
    int d = (e < EE) ? tei[EE + e] : (e - EE);
    atomicAdd(&Af[d * NN + s], dinv[s] * dinv[d]);
  }
  __syncthreads();
  for (int i = tid; i < NN * NN; i += 128) wsA[i] = (_Float16)Af[i];

  // Mt[col][e] = (col<64 ? W_z@L_z_top : W_h@L_h_top)[e][col&63]   (128 x 32)
  for (int o = tid; o < NN * 2 * HID; o += 128) {
    int e = o >> 7, col = o & 127;
    const float* W = (col < HID) ? W_z : W_h;
    const float* L = (col < HID) ? L_z : L_h;
    int c = col & (HID - 1);
    float acc = 0.f;
    for (int k = 0; k < HID; ++k) acc += W[e * HID + k] * L[k * HID + c];
    wsMt[col * NN + e] = (_Float16)acc;   // transposed store: K contiguous per col
  }
  // folded biases
  for (int h = tid; h < HID; h += 128) {
    float az = lb_z[h], ah = lb_h[h];
    for (int k = 0; k < HID; ++k) {
      az += b_z[k] * L_z[k * HID + h];
      ah += b_h[k] * L_h[k * HID + h];
    }
    wscz[h] = az;
    wsch[h] = ah;
  }
  // softmax(att) prefix sums: prefix[k] = sum_{t<k} p_t, k = 0..37
  if (tid == 0) {
    float m = att[0];
    for (int t = 1; t < TT; ++t) m = fmaxf(m, att[t]);
    float p[TT], sum = 0.f;
    for (int t = 0; t < TT; ++t) { p[t] = __expf(att[t] - m); sum += p[t]; }
    float inv = 1.f / sum, run = 0.f;
    for (int k = 0; k <= TT; ++k) { wsprefix[k] = run; if (k < TT) run += p[k] * inv; }
  }
}

// ---------------------------------------------------------------------------
// CDNA5 hardware tanh (VOP1 TRANS op). ISA 7.3: one independent instruction
// (or V_NOP) must follow a TRANS op before its result is consumed / sources
// are overwritten — the asm block embeds the V_NOP itself.
// ---------------------------------------------------------------------------
__device__ __forceinline__ float htanh(float x) {
  float r;
  asm volatile("v_tanh_f32 %0, %1\n\tv_nop" : "=v"(r) : "v"(x));
  return r;
}

// ---------------------------------------------------------------------------
// WMMA fragment helpers (wave32; contiguous K = half*16 + i order in both A and B)
// ---------------------------------------------------------------------------

// A-matrix 16x32 f16: lane holds row r=lane&15; K chunk depends on lane half:
//   a[0..15] = K = half*16 + 0..15 (matches load_bfrag_ct)
// Row-major row pointer -> one contiguous 32B run (2x ds_load_b128).
__device__ __forceinline__ v16h load_afrag(const _Float16* __restrict__ row) {
  const int h16 = ((threadIdx.x >> 4) & 1) * 16;
  v16h a;
#pragma unroll
  for (int i = 0; i < 16; ++i) a[i] = row[h16 + i];
  return a;
}

// B-matrix 32x16 f16 from a K-CONTIGUOUS (column-major) column: lanes 0-15 take
// K=0..15, lanes 16-31 take K=16..31 -> one 32B contiguous run (2x ds_load_b128).
__device__ __forceinline__ v16h load_bfrag_ct(const _Float16* __restrict__ colp) {
  const int kb = ((threadIdx.x >> 4) & 1) * 16;
  v16h b;
#pragma unroll
  for (int i = 0; i < 16; ++i) b[i] = colp[kb + i];
  return b;
}

__device__ __forceinline__ v8f wmma16(v16h a, v16h b, v8f c) {
  return __builtin_amdgcn_wmma_f32_16x16x32_f16(
      /*neg_a=*/false, a, /*neg_b=*/false, b,
      /*c_mod=*/(short)0, c, /*reuse_a=*/false, /*reuse_b=*/false);
}

// ---------------------------------------------------------------------------
// Main kernel: 128 blocks x 128 threads; 1 wave == 1 batch element.
// ---------------------------------------------------------------------------
__global__ __launch_bounds__(128)
void tgcn_main_kernel(const int* __restrict__ x_batch, const int* __restrict__ LOS,
                      const float* __restrict__ emb_table,
                      const float* __restrict__ Wc1, const float* __restrict__ bc1,
                      const float* __restrict__ Wc2, const float* __restrict__ bc2,
                      const _Float16* __restrict__ wsA, const _Float16* __restrict__ wsMt,
                      const float* __restrict__ wscz, const float* __restrict__ wsch,
                      const float* __restrict__ wsprefix,
                      float* __restrict__ out) {
  __shared__ __align__(16) _Float16 sA[NN * NN];        // 2 KB, row-major
  __shared__ __align__(16) _Float16 sMt[2 * HID * NN];  // 8 KB, col-major (K contig)
  __shared__ float scz[HID], sch[HID], sprefix[TT + 1];
  __shared__ float sWc1[HID * 2 * HID];                 // 32 KB
  __shared__ float sbc1[2 * HID], sWc2[2 * HID];
  __shared__ __align__(16) _Float16 sXY[4][NN * EMB];   // per-wave Xt, then Y (8 KB)
  __shared__ float sP[4][HID];                          // per-wave pooled vector

  const int tid = threadIdx.x;
  // ---- stage shared operands (whole block) ----
  for (int i = tid; i < NN * NN; i += 128) sA[i] = wsA[i];
  for (int i = tid; i < NN * 2 * HID; i += 128) sMt[i] = wsMt[i];
  for (int i = tid; i < HID; i += 128) { scz[i] = wscz[i]; sch[i] = wsch[i]; }
  for (int i = tid; i < TT + 1; i += 128) sprefix[i] = wsprefix[i];
  for (int i = tid; i < HID * 2 * HID; i += 128) sWc1[i] = Wc1[i];
  for (int i = tid; i < 2 * HID; i += 128) { sbc1[i] = bc1[i]; sWc2[i] = Wc2[i]; }
  __syncthreads();

  const int w = tid >> 5;
  const int lane = tid & 31;
  const int lane15 = lane & 15;
  const int halfw = (lane >> 4) & 1;
  const int b = blockIdx.x * 4 + w;     // 128 * 4 == 512 batches exactly

  _Float16* XY = &sXY[w][0];

  int li = LOS[b]; if (li > TT) li = TT;
  const float wad = sprefix[li];
  const float wt[2] = { wad, 1.f - wad };
  float acc[4] = {0.f, 0.f, 0.f, 0.f};    // pooled partials, h = nt*16 + lane15

  // ---- hoist stream-invariant fragments into registers ----
  v16h aAdj[2];                 // adjacency A-fragments (GEMM1)
#pragma unroll
  for (int mt = 0; mt < 2; ++mt)
    aAdj[mt] = load_afrag(sA + (mt * 16 + lane15) * NN);

  v16h mBz[4], mBh[4];          // M B-fragments (GEMM2), K-contiguous loads
#pragma unroll
  for (int nt = 0; nt < 4; ++nt) {
    const int col = nt * 16 + lane15;
    mBz[nt] = load_bfrag_ct(sMt + col * NN);
    mBh[nt] = load_bfrag_ct(sMt + (col + HID) * NN);
  }

  for (int s = 0; s < 2; ++s) {
    // ---- gather node embeddings COLUMN-MAJOR: Xt[emb][node=lane] ----
    // (scatter stores, but makes both GEMM1 B-fragments contiguous b128 loads)
    {
      const int c = lane + s * NN;                 // channel (ad: 0..31, dis: 32..63)
      const int idx = x_batch[b * CC + c];
      const float* src = emb_table + ((size_t)c * VV + (size_t)idx) * EMB;
#pragma unroll
      for (int e4 = 0; e4 < 8; ++e4) {
        float4 v = reinterpret_cast<const float4*>(src)[e4];
        XY[(e4 * 4 + 0) * NN + lane] = (_Float16)v.x;
        XY[(e4 * 4 + 1) * NN + lane] = (_Float16)v.y;
        XY[(e4 * 4 + 2) * NN + lane] = (_Float16)v.z;
        XY[(e4 * 4 + 3) * NN + lane] = (_Float16)v.w;
      }
    }
    __syncthreads();

    // ---- GEMM1: Y(32x32) = A(32x32) @ X(32x32); B-frag col = emb index ----
    v16h bx0 = load_bfrag_ct(XY + lane15 * NN);
    v16h bx1 = load_bfrag_ct(XY + (16 + lane15) * NN);
    v8f zero = {};
    v8f Yd[2][2];
#pragma unroll
    for (int mt = 0; mt < 2; ++mt) {
      Yd[mt][0] = wmma16(aAdj[mt], bx0, zero);
      Yd[mt][1] = wmma16(aAdj[mt], bx1, zero);
    }
    __syncthreads();
    // store Y (f16) ROW-major over the same buffer;
    // D layout: row = half*8 + v + mt*16, col = nt*16 + lane15
#pragma unroll
    for (int mt = 0; mt < 2; ++mt)
#pragma unroll
      for (int nt = 0; nt < 2; ++nt)
#pragma unroll
        for (int v = 0; v < 8; ++v)
          XY[(halfw * 8 + v + mt * 16) * EMB + nt * 16 + lane15] =
              (_Float16)Yd[mt][nt][v];
    __syncthreads();

    // ---- GEMM2: G(32x128) = Y @ M ; Gz = cols 0..63, Gh = cols 64..127 ----
    // cell = (1 - sigmoid(Gz + cz)) * tanh(Gh + ch)
    //      = (0.5 - 0.5*tanh(0.5*(Gz + cz))) * tanh(Gh + ch)
#pragma unroll
    for (int mt = 0; mt < 2; ++mt) {
      v16h ya = load_afrag(XY + (mt * 16 + lane15) * EMB);
#pragma unroll
      for (int nt = 0; nt < 4; ++nt) {
        const int col = nt * 16 + lane15;
        v8f gz = wmma16(ya, mBz[nt], zero);
        v8f gh = wmma16(ya, mBh[nt], zero);
        const float czl = scz[col], chl = sch[col];
        float ssum = 0.f;
#pragma unroll
        for (int v = 0; v < 8; ++v) {
          float tz = htanh(0.5f * (gz[v] + czl));   // v_tanh_f32
          float th = htanh(gh[v] + chl);            // v_tanh_f32
          ssum += (0.5f - 0.5f * tz) * th;
        }
        acc[nt] += wt[s] * ssum;
      }
    }
    __syncthreads();
  }

  // ---- pool over nodes (sum over v done; fold the two lane halves) ----
#pragma unroll
  for (int nt = 0; nt < 4; ++nt) {
    float t = acc[nt] + __shfl_xor(acc[nt], 16);
    if (lane < 16) sP[w][nt * 16 + lane] = t * (1.f / (float)NN);
  }
  __syncthreads();

  // ---- head: relu(pooled @ Wc1 + bc1) @ Wc2 + bc2 ----
  float partial = 0.f;
#pragma unroll
  for (int q = 0; q < 4; ++q) {
    const int j = lane + q * 32;          // 128 hidden units over 32 lanes
    float hj = sbc1[j];
#pragma unroll 8
    for (int h = 0; h < HID; ++h) hj += sP[w][h] * sWc1[h * 2 * HID + j];
    hj = fmaxf(hj, 0.f);
    partial += hj * sWc2[j];
  }
#pragma unroll
  for (int off = 16; off >= 1; off >>= 1) partial += __shfl_xor(partial, off);
  if (lane == 0) out[b] = partial + bc2[0];
}

// ---------------------------------------------------------------------------
// Host-side launcher
// ---------------------------------------------------------------------------
extern "C" void kernel_launch(void* const* d_in, const int* in_sizes, int n_in,
                              void* d_out, int out_size, void* d_ws, size_t ws_size,
                              hipStream_t stream) {
  (void)in_sizes; (void)n_in; (void)out_size; (void)ws_size;

  const int*   x_batch = (const int*)  d_in[0];
  const int*   LOS     = (const int*)  d_in[1];
  const int*   tei     = (const int*)  d_in[2];
  const float* emb     = (const float*)d_in[3];
  const float* W_z     = (const float*)d_in[4];
  const float* b_z     = (const float*)d_in[5];
  // d_in[6] = W_r, d_in[7] = b_r, d_in[12] = L_r, d_in[13] = lb_r are dead:
  // H0 == 0 makes the reset-gate path contribute nothing to the output.
  const float* W_h     = (const float*)d_in[8];
  const float* b_h     = (const float*)d_in[9];
  const float* L_z     = (const float*)d_in[10];
  const float* lb_z    = (const float*)d_in[11];
  const float* L_h     = (const float*)d_in[14];
  const float* lb_h    = (const float*)d_in[15];
  const float* att     = (const float*)d_in[16];
  const float* Wc1     = (const float*)d_in[17];
  const float* bc1     = (const float*)d_in[18];
  const float* Wc2     = (const float*)d_in[19];
  const float* bc2     = (const float*)d_in[20];

  // workspace carve-up
  _Float16* wsA      = (_Float16*)d_ws;                 // 1024 halfs
  _Float16* wsMt     = wsA + NN * NN;                   // 4096 halfs (transposed M)
  float*    wscz     = (float*)(wsMt + NN * 2 * HID);   // 64 f32
  float*    wsch     = wscz + HID;                      // 64 f32
  float*    wsprefix = wsch + HID;                      // 38 f32

  tgcn_setup_kernel<<<1, 128, 0, stream>>>(
      tei, W_z, b_z, W_h, b_h, L_z, lb_z, L_h, lb_h, att,
      wsA, wsMt, wscz, wsch, wsprefix);

  tgcn_main_kernel<<<BB / 4, 128, 0, stream>>>(
      x_batch, LOS, emb, Wc1, bc1, Wc2, bc2,
      wsA, wsMt, wscz, wsch, wsprefix, (float*)d_out);
}
